// SetAbstraction_30657476559327
// MI455X (gfx1250) — matrix operand order repaired and checked by reading, not compiled, not run on hardware
//
#include <hip/hip_runtime.h>

typedef _Float16 v16h __attribute__((ext_vector_type(16)));
typedef float    v8f  __attribute__((ext_vector_type(8)));

#define B_    16
#define N_    4096
#define S_    1024
#define K_    32
#define CF_   64
#define CIN0_ 67
#define ROWS_ (B_ * S_ * K_)   // 524288
#define EPS_  1e-5f

// ---------------------------------------------------------------------------
// 1) Farthest point sampling: one block per batch, whole cloud in LDS.
//    Emits new_xyz directly into d_out.
// ---------------------------------------------------------------------------
__global__ __launch_bounds__(256) void fps_kernel(const float* __restrict__ xyz,
                                                  float* __restrict__ new_xyz) {
    __shared__ float lxyz[N_ * 3];
    __shared__ float sval[256];
    __shared__ int   sidx[256];
    __shared__ int   lfar[S_];
    const int b = blockIdx.x, tid = threadIdx.x;
    const float* xb = xyz + (size_t)b * N_ * 3;
    for (int i = tid; i < N_ * 3; i += 256) lxyz[i] = xb[i];
    float dist[16];
#pragma unroll
    for (int j = 0; j < 16; ++j) dist[j] = 1e10f;
    __syncthreads();

    int far = 0;
    for (int it = 0; it < S_; ++it) {
        if (tid == 0) lfar[it] = far;
        const float cx = lxyz[far * 3 + 0];
        const float cy = lxyz[far * 3 + 1];
        const float cz = lxyz[far * 3 + 2];
        float best = -1.0f; int bidx = 0;
#pragma unroll
        for (int j = 0; j < 16; ++j) {
            const int p = tid + j * 256;
            const float dx = lxyz[p * 3 + 0] - cx;
            const float dy = lxyz[p * 3 + 1] - cy;
            const float dz = lxyz[p * 3 + 2] - cz;
            const float d  = dx * dx + dy * dy + dz * dz;
            const float nd = fminf(dist[j], d);
            dist[j] = nd;
            if (nd > best) { best = nd; bidx = p; }
        }
        sval[tid] = best; sidx[tid] = bidx;
        __syncthreads();
        for (int off = 128; off > 0; off >>= 1) {
            if (tid < off) {
                const float v2 = sval[tid + off]; const int i2 = sidx[tid + off];
                if (v2 > sval[tid] || (v2 == sval[tid] && i2 < sidx[tid])) {
                    sval[tid] = v2; sidx[tid] = i2;
                }
            }
            __syncthreads();
        }
        far = sidx[0];
        __syncthreads();
    }
    __syncthreads();
    for (int s = tid; s < S_; s += 256) {
        const int p = lfar[s];
        new_xyz[((size_t)b * S_ + s) * 3 + 0] = lxyz[p * 3 + 0];
        new_xyz[((size_t)b * S_ + s) * 3 + 1] = lxyz[p * 3 + 1];
        new_xyz[((size_t)b * S_ + s) * 3 + 2] = lxyz[p * 3 + 2];
    }
}

// ---------------------------------------------------------------------------
// 2) 32-NN per query: one block per (b,s); d2 table in LDS, 32x argmin.
// ---------------------------------------------------------------------------
__global__ __launch_bounds__(128) void knn_kernel(const float* __restrict__ xyz,
                                                  const float* __restrict__ new_xyz,
                                                  int* __restrict__ knn_idx) {
    __shared__ float d2[N_];
    __shared__ float sval[128];
    __shared__ int   sidx[128];
    const int q = blockIdx.x;          // b*S_ + s
    const int b = q >> 10;
    const int tid = threadIdx.x;
    const float qx = new_xyz[(size_t)q * 3 + 0];
    const float qy = new_xyz[(size_t)q * 3 + 1];
    const float qz = new_xyz[(size_t)q * 3 + 2];
    const float* xb = xyz + (size_t)b * N_ * 3;
    for (int p = tid; p < N_; p += 128) {
        const float dx = xb[p * 3 + 0] - qx;
        const float dy = xb[p * 3 + 1] - qy;
        const float dz = xb[p * 3 + 2] - qz;
        d2[p] = dx * dx + dy * dy + dz * dz;
    }
    __syncthreads();
    for (int j = 0; j < K_; ++j) {
        float best = 3.4e38f; int bidx = 0;
#pragma unroll
        for (int jj = 0; jj < N_ / 128; ++jj) {
            const int p = tid + jj * 128;
            const float v = d2[p];
            if (v < best) { best = v; bidx = p; }
        }
        sval[tid] = best; sidx[tid] = bidx;
        __syncthreads();
        for (int off = 64; off > 0; off >>= 1) {
            if (tid < off) {
                const float v2 = sval[tid + off]; const int i2 = sidx[tid + off];
                if (v2 < sval[tid] || (v2 == sval[tid] && i2 < sidx[tid])) {
                    sval[tid] = v2; sidx[tid] = i2;
                }
            }
            __syncthreads();
        }
        if (tid == 0) {
            knn_idx[(size_t)q * K_ + j] = sidx[0];
            d2[sidx[0]] = 3.4e38f;
        }
        __syncthreads();
    }
}

// ---------------------------------------------------------------------------
// 3) Layer 0: fused gather (xyz-centroid | points) + WMMA GEMM 67(->96 pad)->64
//    256 threads = 8 waves; each wave owns a 16-row tile, full 64 outputs.
// ---------------------------------------------------------------------------
__global__ __launch_bounds__(256) void gemm0_kernel(const float* __restrict__ xyz,
                                                    const float* __restrict__ points,
                                                    const float* __restrict__ new_xyz,
                                                    const int* __restrict__ knn_idx,
                                                    const float* __restrict__ w0,
                                                    const float* __restrict__ bias0,
                                                    _Float16* __restrict__ act1) {
    constexpr int CINP = 96, STRA = 100, COUT = 64;
    __shared__ _Float16 hA[128 * STRA];
    __shared__ _Float16 wL[COUT * CINP];
    const int tid = threadIdx.x;
    const size_t rowBase = (size_t)blockIdx.x * 128;

    for (int i = tid; i < COUT * CINP; i += 256) {
        const int n = i / CINP, k = i % CINP;
        wL[i] = (k < CIN0_) ? (_Float16)w0[n * CIN0_ + k] : (_Float16)0.0f;
    }
    {   // gather: 2 threads per row, 48 channels each
        const int r = tid >> 1, half = tid & 1;
        const size_t gid = rowBase + r;
        const int b = (int)(gid >> 15);
        const int s = (int)((gid >> 5) & 1023);
        const int pt = knn_idx[gid];
        const float* nx = &new_xyz[((size_t)b * S_ + s) * 3];
        const float* xp = &xyz[((size_t)b * N_ + pt) * 3];
        const float* pp = &points[((size_t)b * N_ + pt) * CF_];
        for (int c = half * 48; c < half * 48 + 48; ++c) {
            float v;
            if (c < 3)            v = xp[c] - nx[c];
            else if (c < CIN0_)   v = pp[c - 3];
            else                  v = 0.0f;
            hA[r * STRA + c] = (_Float16)v;
        }
    }
    __syncthreads();

    const int wave = tid >> 5, lane = tid & 31;
    const int M = lane & 15, hi = lane >> 4;
    const int kb = hi * 8, kb2 = hi * 16;
    v8f acc[4];
    const v8f zero = {0.f, 0.f, 0.f, 0.f, 0.f, 0.f, 0.f, 0.f};
#pragma unroll
    for (int nt = 0; nt < 4; ++nt) acc[nt] = zero;

#pragma unroll
    for (int kc = 0; kc < 3; ++kc) {
        v16h a;
        const int abase = (wave * 16 + M) * STRA + kc * 32;
#pragma unroll
        for (int j = 0; j < 8; ++j) {
            a[j]     = hA[abase + kb2 + j];
            a[8 + j] = hA[abase + kb2 + 8 + j];
        }
#pragma unroll
        for (int nt = 0; nt < 4; ++nt) {
            v16h bf;
            const int bbase = (nt * 16 + M) * CINP + kc * 32 + kb2;
#pragma unroll
            for (int j = 0; j < 16; ++j) bf[j] = wL[bbase + j];
            acc[nt] = __builtin_amdgcn_wmma_f32_16x16x32_f16(
                false, a, false, bf, (short)0, acc[nt], false, false);
        }
    }
#pragma unroll
    for (int nt = 0; nt < 4; ++nt) {
        const int col = nt * 16 + M;
        const float bb = bias0[col];
#pragma unroll
        for (int j = 0; j < 8; ++j) {
            const int row = wave * 16 + j + hi * 8;
            act1[(rowBase + row) * (size_t)COUT + col] = (_Float16)(acc[nt][j] + bb);
        }
    }
}

// ---------------------------------------------------------------------------
// 4) Layers 1/2: BN+ReLU on input (prev-layer stats) + WMMA GEMM.
//    KCH = #K-chunks of 32, NT = #16-wide N tiles.
// ---------------------------------------------------------------------------
template <int KCH, int NT>
__global__ __launch_bounds__(256) void gemm_bn_kernel(const _Float16* __restrict__ actIn,
                                                      const float* __restrict__ w,
                                                      const float* __restrict__ bias,
                                                      const float* __restrict__ gamma,
                                                      const float* __restrict__ beta,
                                                      const float* __restrict__ mv,
                                                      _Float16* __restrict__ actOut) {
    constexpr int CIN = KCH * 32, COUT = NT * 16, STRA = CIN + 4;
    __shared__ _Float16 hA[128 * STRA];
    __shared__ _Float16 wL[COUT * CIN];
    __shared__ float sm[CIN], si[CIN], sg[CIN], sb[CIN];
    const int tid = threadIdx.x;
    const size_t rowBase = (size_t)blockIdx.x * 128;

    if (tid < CIN) {
        sm[tid] = mv[tid];
        si[tid] = mv[CIN + tid];
        sg[tid] = gamma[tid];
        sb[tid] = beta[tid];
    }
    for (int i = tid; i < COUT * CIN; i += 256) wL[i] = (_Float16)w[i];
    __syncthreads();
    for (int i = tid; i < 128 * CIN; i += 256) {
        const int r = i / CIN, k = i % CIN;
        float x = (float)actIn[(rowBase + r) * (size_t)CIN + k];
        x = fmaxf((x - sm[k]) * si[k] * sg[k] + sb[k], 0.0f);
        hA[r * STRA + k] = (_Float16)x;
    }
    __syncthreads();

    const int wave = tid >> 5, lane = tid & 31;
    const int M = lane & 15, hi = lane >> 4;
    const int kb = hi * 8, kb2 = hi * 16;
    v8f acc[NT];
    const v8f zero = {0.f, 0.f, 0.f, 0.f, 0.f, 0.f, 0.f, 0.f};
#pragma unroll
    for (int nt = 0; nt < NT; ++nt) acc[nt] = zero;

#pragma unroll
    for (int kc = 0; kc < KCH; ++kc) {
        v16h a;
        const int abase = (wave * 16 + M) * STRA + kc * 32;
#pragma unroll
        for (int j = 0; j < 8; ++j) {
            a[j]     = hA[abase + kb2 + j];
            a[8 + j] = hA[abase + kb2 + 8 + j];
        }
#pragma unroll
        for (int nt = 0; nt < NT; ++nt) {
            v16h bf;
            const int bbase = (nt * 16 + M) * CIN + kc * 32 + kb2;
#pragma unroll
            for (int j = 0; j < 16; ++j) bf[j] = wL[bbase + j];
            acc[nt] = __builtin_amdgcn_wmma_f32_16x16x32_f16(
                false, a, false, bf, (short)0, acc[nt], false, false);
        }
    }
#pragma unroll
    for (int nt = 0; nt < NT; ++nt) {
        const int col = nt * 16 + M;
        const float bb = bias[col];
#pragma unroll
        for (int j = 0; j < 8; ++j) {
            const int row = wave * 16 + j + hi * 8;
            actOut[(rowBase + row) * (size_t)COUT + col] = (_Float16)(acc[nt][j] + bb);
        }
    }
}

// ---------------------------------------------------------------------------
// 5) Deterministic per-channel sum / sumsq (two-stage, no float atomics).
// ---------------------------------------------------------------------------
template <int C>
__global__ __launch_bounds__(256) void stats1_kernel(const _Float16* __restrict__ act,
                                                     float* __restrict__ partial) {
    constexpr int RPB = ROWS_ / 256;     // 2048 rows per block
    constexpr int G   = 256 / C;         // row-lanes per channel
    const int tid = threadIdx.x;
    const int c = tid % C, g = tid / C;
    const size_t r0 = (size_t)blockIdx.x * RPB;
    float s = 0.0f, sq = 0.0f;
    for (int r = g; r < RPB; r += G) {
        const float x = (float)act[(r0 + r) * (size_t)C + c];
        s += x; sq += x * x;
    }
    __shared__ float ss[256], sqq[256];
    ss[tid] = s; sqq[tid] = sq;
    __syncthreads();
    if (g == 0) {
        for (int gg = 1; gg < G; ++gg) { s += ss[gg * C + c]; sq += sqq[gg * C + c]; }
        partial[blockIdx.x * 2 * C + c]     = s;
        partial[blockIdx.x * 2 * C + C + c] = sq;
    }
}

template <int C>
__global__ void stats2_kernel(const float* __restrict__ partial, float* __restrict__ mv) {
    const int c = threadIdx.x;
    float s = 0.0f, sq = 0.0f;
    for (int b = 0; b < 256; ++b) {
        s  += partial[b * 2 * C + c];
        sq += partial[b * 2 * C + C + c];
    }
    const float inv_n = 1.0f / (float)ROWS_;
    const float mean = s * inv_n;
    float var = sq * inv_n - mean * mean;
    var = fmaxf(var, 0.0f);
    mv[c]     = mean;
    mv[C + c] = rsqrtf(var + EPS_);
}

// ---------------------------------------------------------------------------
// 6) BN+ReLU on act3, max over K, transposed store (B, 128, S).
// ---------------------------------------------------------------------------
__global__ __launch_bounds__(128) void final_kernel(const _Float16* __restrict__ act3,
                                                    const float* __restrict__ gamma,
                                                    const float* __restrict__ beta,
                                                    const float* __restrict__ mv,
                                                    float* __restrict__ out_points) {
    const int q = blockIdx.x;            // b*S_ + s
    const int c = threadIdx.x;           // channel 0..127
    const int b = q >> 10, s = q & 1023;
    const float mean = mv[c], inv = mv[128 + c];
    const float gg = gamma[c], bb = beta[c];
    float m = -3.4e38f;
    const size_t base = (size_t)q * K_ * 128 + c;
#pragma unroll 4
    for (int k = 0; k < K_; ++k) {
        float x = (float)act3[base + (size_t)k * 128];
        x = fmaxf((x - mean) * inv * gg + bb, 0.0f);
        m = fmaxf(m, x);
    }
    out_points[((size_t)b * 128 + c) * S_ + s] = m;
}

// ---------------------------------------------------------------------------
// Launch
// ---------------------------------------------------------------------------
extern "C" void kernel_launch(void* const* d_in, const int* in_sizes, int n_in,
                              void* d_out, int out_size, void* d_ws, size_t ws_size,
                              hipStream_t stream) {
    const float* xyz    = (const float*)d_in[0];
    const float* points = (const float*)d_in[1];
    const float* w0  = (const float*)d_in[2];
    const float* b0  = (const float*)d_in[3];
    const float* g0  = (const float*)d_in[4];
    const float* bb0 = (const float*)d_in[5];
    const float* w1  = (const float*)d_in[6];
    const float* b1  = (const float*)d_in[7];
    const float* g1  = (const float*)d_in[8];
    const float* bb1 = (const float*)d_in[9];
    const float* w2  = (const float*)d_in[10];
    const float* b2  = (const float*)d_in[11];
    const float* g2  = (const float*)d_in[12];
    const float* bb2 = (const float*)d_in[13];

    float* out        = (float*)d_out;
    float* new_xyz    = out;                         // B*S*3 = 49152
    float* new_points = out + (size_t)B_ * S_ * 3;   // B*128*S

    char* ws = (char*)d_ws;
    constexpr size_t MB = 1ull << 20;
    constexpr size_t OFF_IDX  = 0;                 // 2 MB: knn indices
    constexpr size_t OFF_MV   = 2 * MB;            // mean/invstd for 3 layers
    constexpr size_t OFF_PART = 2 * MB + 65536;    // 256 KB: stage-1 partials
    constexpr size_t OFF_R1   = 4 * MB;            // 128 MB: act1 then act3
    constexpr size_t OFF_R2   = 132 * MB;          // 64 MB: act2

    int*      knn_idx = (int*)(ws + OFF_IDX);
    float*    mv0     = (float*)(ws + OFF_MV);     // 128 floats
    float*    mv1     = mv0 + 128;                 // 128 floats
    float*    mv2     = mv1 + 128;                 // 256 floats
    float*    partial = (float*)(ws + OFF_PART);
    _Float16* act1    = (_Float16*)(ws + OFF_R1);
    _Float16* act3    = (_Float16*)(ws + OFF_R1);  // reuses act1's region
    _Float16* act2    = (_Float16*)(ws + OFF_R2);

    fps_kernel<<<B_, 256, 0, stream>>>(xyz, new_xyz);
    knn_kernel<<<B_ * S_, 128, 0, stream>>>(xyz, new_xyz, knn_idx);

    gemm0_kernel<<<ROWS_ / 128, 256, 0, stream>>>(xyz, points, new_xyz, knn_idx,
                                                  w0, b0, act1);
    stats1_kernel<64><<<256, 256, 0, stream>>>(act1, partial);
    stats2_kernel<64><<<1, 64, 0, stream>>>(partial, mv0);

    gemm_bn_kernel<2, 4><<<ROWS_ / 128, 256, 0, stream>>>(act1, w1, b1, g0, bb0,
                                                          mv0, act2);
    stats1_kernel<64><<<256, 256, 0, stream>>>(act2, partial);
    stats2_kernel<64><<<1, 64, 0, stream>>>(partial, mv1);

    gemm_bn_kernel<2, 8><<<ROWS_ / 128, 256, 0, stream>>>(act2, w2, b2, g1, bb1,
                                                          mv1, act3);
    stats1_kernel<128><<<256, 256, 0, stream>>>(act3, partial);
    stats2_kernel<128><<<1, 128, 0, stream>>>(partial, mv2);

    final_kernel<<<B_ * S_, 128, 0, stream>>>(act3, g2, bb2, mv2, new_points);
}
